// KnowledgeEncoder_29240137351560
// MI455X (gfx1250) — matrix operand order repaired and checked by reading, not compiled, not run on hardware
//
#include <hip/hip_runtime.h>
#include <hip/hip_bf16.h>

// ---------------- problem constants (match reference) ----------------
#define L_    8
#define J_    3
#define N_    2000
#define E_    32000
#define R_    64
#define HID_  1024
#define G_    128
#define H_    8
#define DH_   16
#define P_    10
#define OUT2_ 512       // 2*OUT
#define B_    16
#define KV_   8
#define NR_   (N_ + R_) // 2064 combined rows (nodes ++ relations), 129 M-tiles
#define MT_   4         // M-subtiles per edge block (64 edges/block)

#define TGW_ELEMS (L_ * HID_ * G_)          // 1,048,576 bf16
#define PW_ELEMS  (L_ * J_ * 3 * G_ * G_)   // 1,179,648 bf16

typedef __attribute__((ext_vector_type(16))) __bf16 v16bf;
typedef __attribute__((ext_vector_type(8)))  __bf16 bf16x8;
typedef __attribute__((ext_vector_type(8)))  float  v8f;
typedef __attribute__((ext_vector_type(4)))  float  f32x4;

// D = A(16x32 bf16) * B(32x16 bf16) + C(16x16 f32)   [wave32]
__device__ __forceinline__ v8f wmma_bf16(v16bf a, v16bf b, v8f c) {
    return __builtin_amdgcn_wmma_f32_16x16x32_bf16(
        false, a, false, b, (short)0, c, false, false);
}

__device__ __forceinline__ v16bf cat16(bf16x8 lo, bf16x8 hi) {
    return __builtin_shufflevector(lo, hi, 0, 1, 2, 3, 4, 5, 6, 7,
                                   8, 9, 10, 11, 12, 13, 14, 15);
}

__device__ __forceinline__ bf16x8 cvt8(f32x4 x0, f32x4 x1) {
    bf16x8 o;
    o[0] = (__bf16)x0[0]; o[1] = (__bf16)x0[1];
    o[2] = (__bf16)x0[2]; o[3] = (__bf16)x0[3];
    o[4] = (__bf16)x1[0]; o[5] = (__bf16)x1[1];
    o[6] = (__bf16)x1[2]; o[7] = (__bf16)x1[3];
    return o;
}

// ---------------------------------------------------------------------
// Pack weights to bf16, fragment-major: [ktile][col][32 k-elems] so a
// lane's 16 B-fragment elements are two contiguous 16B global loads.
// ---------------------------------------------------------------------
__global__ __launch_bounds__(256)
void k_pack(const float* __restrict__ tg_w, const float* __restrict__ path_w,
            __bf16* __restrict__ tgwb, __bf16* __restrict__ pwb)
{
    const int idx = blockIdx.x * 256 + threadIdx.x;
    if (idx < TGW_ELEMS) {              // layout ((l*32+kt)*128+col)*32+kk
        const int kk  = idx & 31;
        const int col = (idx >> 5) & 127;
        const int kt  = (idx >> 12) & 31;
        const int l   = idx >> 17;
        tgwb[idx] = (__bf16)tg_w[((size_t)l * HID_ + kt * 32 + kk) * G_ + col];
    }
    if (idx < PW_ELEMS) {               // layout ((lj*12+kt)*128+col)*32+kk
        const int kk  = idx & 31;
        const int col = (idx >> 5) & 127;
        const int kt  = (idx >> 12) % 12;
        const int lj  = (idx >> 12) / 12;
        pwb[idx] = (__bf16)path_w[((size_t)lj * (3 * G_) + kt * 32 + kk) * G_ + col];
    }
}

// ---------------------------------------------------------------------
// CSR-by-destination build (edge_dst is constant across all GAT steps)
// ---------------------------------------------------------------------
__global__ void k_zero(int* __restrict__ cnt, int* __restrict__ cursor)
{
    const int n = blockIdx.x * blockDim.x + threadIdx.x;
    if (n < N_) { cnt[n] = 0; cursor[n] = 0; }
}

__global__ void k_hist(const int* __restrict__ edst, int* __restrict__ cnt)
{
    const int e = blockIdx.x * blockDim.x + threadIdx.x;
    if (e < E_) atomicAdd(&cnt[edst[e]], 1);
}

// exclusive scan of cnt[0..N) -> off[0..N], single block of 256
__global__ __launch_bounds__(256)
void k_scan(const int* __restrict__ cnt, int* __restrict__ off)
{
    __shared__ int part[256];
    const int tid  = threadIdx.x;
    const int base = tid * 8;
    int loc[8];
    int s = 0;
    #pragma unroll
    for (int q = 0; q < 8; ++q) {
        const int idx = base + q;
        const int c   = (idx < N_) ? cnt[idx] : 0;
        loc[q] = s; s += c;
    }
    part[tid] = s;
    __syncthreads();
    if (tid == 0) {
        int run = 0;
        for (int t = 0; t < 256; ++t) { const int tmp = part[t]; part[t] = run; run += tmp; }
    }
    __syncthreads();
    const int p0 = part[tid];
    #pragma unroll
    for (int q = 0; q < 8; ++q) {
        const int idx = base + q;
        if (idx < N_) off[idx] = p0 + loc[q];
    }
    if (tid == 0) off[N_] = E_;
}

__global__ void k_scatter(const int* __restrict__ edst, const int* __restrict__ off,
                          int* __restrict__ cursor, int* __restrict__ eidx)
{
    const int e = blockIdx.x * blockDim.x + threadIdx.x;
    if (e >= E_) return;
    const int d = edst[e];
    const int slot = atomicAdd(&cursor[d], 1);
    eidx[off[d] + slot] = e;
}

// per-bucket ascending insertion sort -> deterministic accumulation order
__global__ void k_sortb(const int* __restrict__ off, int* __restrict__ eidx)
{
    const int n = blockIdx.x * blockDim.x + threadIdx.x;
    if (n >= N_) return;
    const int o0 = off[n];
    const int d  = off[n + 1] - o0;
    for (int i = 1; i < d; ++i) {
        const int v = eidx[o0 + i];
        int k = i - 1;
        while (k >= 0 && eidx[o0 + k] > v) { eidx[o0 + k + 1] = eidx[o0 + k]; --k; }
        eidx[o0 + k + 1] = v;
    }
}

// ---------------------------------------------------------------------
// x/rel transform: rows [0,2000) = nft[i] @ tg_w + b ; rows [2000,2064) =
// rft[i] @ tg_w + b.  One block = 16 rows, 8 waves = 8 N-tiles of 16 cols.
// ---------------------------------------------------------------------
__global__ __launch_bounds__(256)
void k_transform(const float* __restrict__ nft, const float* __restrict__ rft,
                 const __bf16* __restrict__ tgwb, const float* __restrict__ tg_b,
                 float* __restrict__ xbuf, int layer)
{
    __shared__ __bf16 As[16][HID_];               // 32 KB
    const int tid  = threadIdx.x;
    const int row0 = blockIdx.x * 16;

    { // cooperative fp32->bf16 stage of the 16x1024 A block (streamed once)
        const int r   = tid >> 4;                 // row 0..15
        const int sub = tid & 15;                 // 64 contiguous elems each
        const int row = row0 + r;
        const f32x4* src = (const f32x4*)((row < N_)
            ? (nft + (size_t)layer * N_ * HID_ + (size_t)row * HID_)
            : (rft + (size_t)layer * R_ * HID_ + (size_t)(row - N_) * HID_));
        #pragma unroll
        for (int q = 0; q < 8; ++q) {
            const f32x4 x0 = __builtin_nontemporal_load(&src[sub * 16 + 2 * q]);
            const f32x4 x1 = __builtin_nontemporal_load(&src[sub * 16 + 2 * q + 1]);
            *(bf16x8*)&As[r][sub * 64 + q * 8] = cvt8(x0, x1);
        }
    }
    __syncthreads();

    const int lane = tid & 31;
    const int wid  = tid >> 5;                    // N-tile
    const int m    = lane & 15;
    const int hi   = lane >> 4;
    const int col  = wid * 16 + (lane & 15);
    const __bf16* Bb = tgwb + (((size_t)layer * 32) * G_ + col) * 32 + hi * 16;

    v8f c = {};
    for (int kt = 0; kt < 32; ++kt) {
        const __bf16* bp = Bb + (size_t)kt * (G_ * 32);
        const bf16x8 b0 = *(const bf16x8*)bp;
        const bf16x8 b1 = *(const bf16x8*)(bp + 8);
        if (kt < 31) __builtin_prefetch(bp + G_ * 32, 0, 3);   // WGP-scope
        const bf16x8 a0 = *(const bf16x8*)&As[m][kt * 32 + hi * 16];
        const bf16x8 a1 = *(const bf16x8*)&As[m][kt * 32 + hi * 16 + 8];
        c = wmma_bf16(cat16(a0, a1), cat16(b0, b1), c);
    }
    const float bias = tg_b[layer * G_ + col];
    #pragma unroll
    for (int r = 0; r < 8; ++r)                   // C: M = r + 8*hi, N = col
        xbuf[(size_t)(row0 + r + 8 * hi) * G_ + col] = c[r] + bias;
}

// ---------------------------------------------------------------------
// Per-edge GEMM + attention logits.  Block = 64 edges (4 M-subtiles);
// A = concat(x[src], rel[lbl], x[dst]) staged once in LDS (bf16); each
// wave owns one head's 16-column tile; one B fragment feeds 4 WMMAs.
// ---------------------------------------------------------------------
__global__ __launch_bounds__(256)
void k_edge(const float* __restrict__ xbuf,
            const int* __restrict__ esrc, const int* __restrict__ edst,
            const int* __restrict__ elbl,
            const __bf16* __restrict__ pwb, const float* __restrict__ path_b,
            const float* __restrict__ attn1_w, const float* __restrict__ attn2_p,
            float* __restrict__ epaths, float* __restrict__ abuf,
            int layer, int gj)
{
    __shared__ __bf16 Es[MT_ * 16][3 * G_];       // 48 KB
    __shared__ float  a2s[MT_ * 16][H_];          // 2 KB
    __shared__ float  W1s[H_ * G_];               // 4 KB, transposed [h][k]
    const int tid = threadIdx.x;
    const int e0  = blockIdx.x * (MT_ * 16);
    const int lj  = layer * J_ + gj;

    { // stage 64 x 384 gathered A rows; 8-float chunks never cross segments
        const int r   = tid >> 2;                 // row 0..63
        const int sub = tid & 3;                  // 96 elems each
        const int eg  = e0 + r;
        const float* ps = xbuf + (size_t)esrc[eg] * G_;
        const float* pr = xbuf + (size_t)(N_ + elbl[eg]) * G_;
        const float* pd = xbuf + (size_t)edst[eg] * G_;
        #pragma unroll
        for (int q = 0; q < 12; ++q) {
            const int k = sub * 96 + q * 8;
            const float* base = (k < G_) ? (ps + k)
                              : (k < 2 * G_) ? (pr + k - G_) : (pd + k - 2 * G_);
            const f32x4 x0 = *(const f32x4*)base;
            const f32x4 x1 = *(const f32x4*)(base + 4);
            *(bf16x8*)&Es[r][k] = cvt8(x0, x1);
        }
        const float* W1 = attn1_w + (size_t)lj * G_ * H_;   // [k][h]
        #pragma unroll
        for (int q = 0; q < 4; ++q) {
            const int idx = tid * 4 + q;                    // transpose to [h][k]
            W1s[idx] = W1[(idx & 127) * H_ + (idx >> 7)];
        }
    }
    __syncthreads();

    const int lane = tid & 31;
    const int wid  = tid >> 5;                    // == head h (DH == 16)
    const int m    = lane & 15;
    const int hi   = lane >> 4;
    const int n    = lane & 15;
    const int col  = wid * 16 + n;
    const __bf16* Bb = pwb + (((size_t)lj * 12) * G_ + col) * 32 + hi * 16;
    const float  bp = path_b[lj * G_ + col];
    const float  cf = attn2_p[lj * G_ + col];     // [H,DH] flat == col

    v8f c[MT_] = {};
    for (int kt = 0; kt < 12; ++kt) {
        const __bf16* bpp = Bb + (size_t)kt * (G_ * 32);
        const bf16x8 b0 = *(const bf16x8*)bpp;
        const bf16x8 b1 = *(const bf16x8*)(bpp + 8);
        if (kt < 11) __builtin_prefetch(bpp + G_ * 32, 0, 3);  // WGP-scope
        const v16bf b = cat16(b0, b1);
        #pragma unroll
        for (int mt = 0; mt < MT_; ++mt) {
            const bf16x8 a0 = *(const bf16x8*)&Es[mt * 16 + m][kt * 32 + hi * 16];
            const bf16x8 a1 = *(const bf16x8*)&Es[mt * 16 + m][kt * 32 + hi * 16 + 8];
            c[mt] = wmma_bf16(cat16(a0, a1), b, c[mt]);
        }
    }

    #pragma unroll
    for (int mt = 0; mt < MT_; ++mt) {            // bias, spill epaths, a2 reduce
        float v[8];
        #pragma unroll
        for (int r = 0; r < 8; ++r) {
            const float val = c[mt][r] + bp;
            epaths[(size_t)(e0 + mt * 16 + r + 8 * hi) * G_ + col] = val;
            v[r] = val * cf;
        }
        #pragma unroll
        for (int off = 1; off < 16; off <<= 1) {  // reduce over 16 cols (d)
            #pragma unroll
            for (int r = 0; r < 8; ++r) v[r] += __shfl_xor(v[r], off, 32);
        }
        if (n == 0) {
            #pragma unroll
            for (int r = 0; r < 8; ++r) a2s[mt * 16 + r + 8 * hi][wid] = v[r];
        }
    }
    __syncthreads();

    #pragma unroll
    for (int rep = 0; rep < 2; ++rep) {           // a1 + leaky_relu
        const int idx = rep * 256 + tid;          // 512 = 64 edges x 8 heads
        const int er  = idx >> 3;
        const int h   = idx & 7;
        float s = 0.0f;
        #pragma unroll
        for (int kq = 0; kq < 16; ++kq) {
            const bf16x8 av = *(const bf16x8*)&Es[er][kq * 8];
            const f32x4  w0 = *(const f32x4*)&W1s[h * G_ + kq * 8];
            const f32x4  w1 = *(const f32x4*)&W1s[h * G_ + kq * 8 + 4];
            s += (float)av[0] * w0[0] + (float)av[1] * w0[1]
               + (float)av[2] * w0[2] + (float)av[3] * w0[3]
               + (float)av[4] * w1[0] + (float)av[5] * w1[1]
               + (float)av[6] * w1[2] + (float)av[7] * w1[3];
        }
        float a = s + a2s[er][h];
        a = (a > 0.0f) ? a : 0.01f * a;
        abuf[(e0 + er) * H_ + h] = a;
    }
}

// ---------------------------------------------------------------------
// Edge-softmax + aggregation + relu update, one wave per destination
// node over its (deterministically sorted) CSR edge list.  Lane owns 4
// columns; lanes 0..7 compute per-head max / exp-sum in fixed order.
// ---------------------------------------------------------------------
__global__ __launch_bounds__(256)
void k_node(float* __restrict__ xbuf, const float* __restrict__ abuf,
            const float* __restrict__ epaths,
            const int* __restrict__ off, const int* __restrict__ eidx)
{
    const int tid  = threadIdx.x;
    const int lane = tid & 31;
    const int node = blockIdx.x * 8 + (tid >> 5);   // 250 blocks x 8 waves = N
    const int o0   = off[node];
    const int deg  = off[node + 1] - o0;

    float maxv = -__builtin_inff(), sumv = 0.0f;
    if (lane < H_) {                                // serial, fixed order
        for (int q = 0; q < deg; ++q)
            maxv = fmaxf(maxv, abuf[eidx[o0 + q] * H_ + lane]);
        for (int q = 0; q < deg; ++q)
            sumv += expf(abuf[eidx[o0 + q] * H_ + lane] - maxv);
    }
    const int   hme  = lane >> 2;                   // head of this lane's 4 cols
    const float maxh = __shfl(maxv, hme, 32);
    const float inv  = 1.0f / __shfl(sumv, hme, 32);

    f32x4 acc = {0.f, 0.f, 0.f, 0.f};
    for (int q = 0; q < deg; ++q) {
        const int   e = eidx[o0 + q];
        const float w = expf(abuf[e * H_ + hme] - maxh) * inv;
        const f32x4 ep = *(const f32x4*)(epaths + (size_t)e * G_ + lane * 4);
        #pragma unroll
        for (int t = 0; t < 4; ++t) acc[t] += w * ep[t];
    }
    float* xr = xbuf + (size_t)node * G_ + lane * 4;
    const f32x4 x = *(const f32x4*)xr;
    f32x4 o;
    #pragma unroll
    for (int t = 0; t < 4; ++t) {
        const float v = x[t] + acc[t];
        o[t] = v > 0.0f ? v : 0.0f;
    }
    *(f32x4*)xr = o;
}

// top-10 of importances (layer-invariant): iterative argmax, one block
__global__ __launch_bounds__(256)
void k_topk(const float* __restrict__ imp, int* __restrict__ idxo)
{
    __shared__ float sv[256];
    __shared__ int   si[256];
    __shared__ int   chosen[P_];
    const int tid = threadIdx.x;
    for (int p = 0; p < P_; ++p) {
        float best = -__builtin_inff(); int bi = 0x7fffffff;
        for (int nn = tid; nn < N_; nn += 256) {
            bool used = false;
            for (int q = 0; q < p; ++q) used |= (chosen[q] == nn);
            if (used) continue;
            const float v = imp[nn];
            if (v > best || (v == best && nn < bi)) { best = v; bi = nn; }
        }
        sv[tid] = best; si[tid] = bi;
        __syncthreads();
        for (int s = 128; s > 0; s >>= 1) {
            if (tid < s) {
                if (sv[tid + s] > sv[tid] ||
                    (sv[tid + s] == sv[tid] && si[tid + s] < si[tid])) {
                    sv[tid] = sv[tid + s]; si[tid] = si[tid + s];
                }
            }
            __syncthreads();
        }
        if (tid == 0) { chosen[p] = si[0]; idxo[p] = si[0]; }
        __syncthreads();
    }
}

// tanh(prefix) @ trans_w + trans_b, broadcast over batch, scattered to
// the [L,2,B,KV,P,OUT/KV] output layout
__global__ __launch_bounds__(256)
void k_output(const float* __restrict__ xbuf, const int* __restrict__ idx,
              const float* __restrict__ trans_w, const float* __restrict__ trans_b,
              float* __restrict__ out, int layer)
{
    const int t = blockIdx.x * blockDim.x + threadIdx.x;     // P*512
    if (t >= P_ * OUT2_) return;
    const int p  = t >> 9;
    const int cc = t & (OUT2_ - 1);
    const float* xr = xbuf + (size_t)idx[p] * G_;
    const float* W  = trans_w + (size_t)layer * G_ * OUT2_;
    float s = trans_b[layer * OUT2_ + cc];
    for (int k = 0; k < G_; ++k)
        s += tanhf(xr[k]) * W[k * OUT2_ + cc];
    const int sb = cc >> 8;                                   // 0..1
    const int kv = (cc >> 5) & 7;
    const int o2 = cc & 31;
    for (int b = 0; b < B_; ++b) {
        const size_t o =
            (((((size_t)layer * 2 + sb) * B_ + b) * KV_ + kv) * P_ + p) * 32 + o2;
        out[o] = s;
    }
}

// ---------------------------------------------------------------------
extern "C" void kernel_launch(void* const* d_in, const int* in_sizes, int n_in,
                              void* d_out, int out_size, void* d_ws, size_t ws_size,
                              hipStream_t stream)
{
    (void)in_sizes; (void)n_in; (void)out_size; (void)ws_size;
    const float* nft    = (const float*)d_in[0];
    const float* rft    = (const float*)d_in[1];
    const float* imp    = (const float*)d_in[2];
    const float* tg_w   = (const float*)d_in[3];
    const float* tg_b   = (const float*)d_in[4];
    const float* path_w = (const float*)d_in[5];
    const float* path_b = (const float*)d_in[6];
    const float* a1w    = (const float*)d_in[7];
    const float* a2p    = (const float*)d_in[8];
    const float* tw     = (const float*)d_in[9];
    const float* tb     = (const float*)d_in[10];
    const int*   esrc   = (const int*)d_in[12];
    const int*   edst   = (const int*)d_in[13];
    const int*   elbl   = (const int*)d_in[14];
    float* out = (float*)d_out;

    char* ws = (char*)d_ws;
    float*  xbuf   = (float*)ws;  ws += sizeof(float) * (size_t)NR_ * G_;
    float*  epaths = (float*)ws;  ws += sizeof(float) * (size_t)E_ * G_;
    float*  abuf   = (float*)ws;  ws += sizeof(float) * (size_t)E_ * H_;
    __bf16* tgwb   = (__bf16*)ws; ws += sizeof(__bf16) * (size_t)TGW_ELEMS;
    __bf16* pwb    = (__bf16*)ws; ws += sizeof(__bf16) * (size_t)PW_ELEMS;
    int*    cnt    = (int*)ws;    ws += sizeof(int) * N_;
    int*    cursor = (int*)ws;    ws += sizeof(int) * N_;
    int*    off    = (int*)ws;    ws += sizeof(int) * 2048;   // N_+1, padded
    int*    eidx   = (int*)ws;    ws += sizeof(int) * E_;
    int*    topidx = (int*)ws;

    k_pack<<<(PW_ELEMS + 255) / 256, 256, 0, stream>>>(tg_w, path_w, tgwb, pwb);
    k_topk<<<1, 256, 0, stream>>>(imp, topidx);

    // CSR-by-destination, built once, reused by all 24 GAT steps
    k_zero<<<(N_ + 255) / 256, 256, 0, stream>>>(cnt, cursor);
    k_hist<<<(E_ + 255) / 256, 256, 0, stream>>>(edst, cnt);
    k_scan<<<1, 256, 0, stream>>>(cnt, off);
    k_scatter<<<(E_ + 255) / 256, 256, 0, stream>>>(edst, off, cursor, eidx);
    k_sortb<<<(N_ + 255) / 256, 256, 0, stream>>>(off, eidx);

    for (int i = 0; i < L_; ++i) {
        k_transform<<<NR_ / 16, 256, 0, stream>>>(nft, rft, tgwb, tg_b, xbuf, i);
        for (int j = 0; j < J_; ++j) {
            k_edge<<<E_ / (MT_ * 16), 256, 0, stream>>>(
                xbuf, esrc, edst, elbl, pwb, path_b, a1w, a2p,
                epaths, abuf, i, j);
            k_node<<<N_ / 8, 256, 0, stream>>>(xbuf, abuf, epaths, off, eidx);
        }
        k_output<<<(P_ * OUT2_ + 255) / 256, 256, 0, stream>>>(
            xbuf, topidx, tw, tb, out, i);
    }
    // lamda mixing is an exact no-op in the reference (past_x == x), skipped.
}
